// DecoderLayer_52613349376212
// MI455X (gfx1250) — matrix operand order repaired and checked by reading, not compiled, not run on hardware
//
#include <hip/hip_runtime.h>
#include <math.h>

// ---------- types ----------
typedef __attribute__((ext_vector_type(16))) __bf16 v16bf;
typedef __attribute__((ext_vector_type(8)))  __bf16 v8bf;
typedef __attribute__((ext_vector_type(8)))  float  v8f;
typedef __attribute__((ext_vector_type(4)))  float  v4f;

// epilogue modes
#define EPI_NONE     0
#define EPI_RES      1
#define EPI_BIAS_ELU 2
#define EPI_BIAS_RES 3

// LDS byte-offset of a generic pointer to __shared__ (low 32 bits of the
// LDS-aperture generic address are the LDS offset).
__device__ __forceinline__ unsigned lds_off(const void* p)
{
    return (unsigned)(uintptr_t)p;
}

// Async DMA: 16 bytes memory -> LDS, tracked by ASYNCcnt (no VGPR data).
__device__ __forceinline__ void async_b128_to_lds(unsigned lds_addr, const void* gaddr)
{
    asm volatile("global_load_async_to_lds_b128 %0, %1, off"
                 :: "v"(lds_addr), "v"((unsigned long long)(uintptr_t)gaddr)
                 : "memory");
}

__device__ __forceinline__ void wait_asynccnt0()
{
    asm volatile("s_wait_asynccnt 0x0" ::: "memory");
}

// =====================================================================
// Tiled bf16 WMMA GEMM:  C[M,N] = epi(A[M,K] x Bt[N,K]^T)
// A: bf16 [M,K] row-major. Bt: bf16 [N,K] row-major (pre-transposed B).
// block = 256 threads (8 waves, 4x2), C tile 128x128, 32x64 per wave,
// K-step 32, double-buffered LDS filled by GLOBAL_LOAD_ASYNC_TO_LDS_B128.
// =====================================================================
template <int EPI, bool OUTB>
__global__ void __launch_bounds__(256, 1)
gemm_bf16_wmma(const __bf16* __restrict__ A, const __bf16* __restrict__ Bt,
               void* __restrict__ Cv, const float* __restrict__ bias,
               const float* __restrict__ resid, int M, int N, int K)
{
    __shared__ __align__(16) __bf16 As[2][128][40];  // [buf][row][k], 80B stride
    __shared__ __align__(16) __bf16 Bs[2][128][40];  // [buf][col][k]

    const int t    = threadIdx.x;
    const int lane = t & 31;
    const int wave = t >> 5;
    const int g    = lane >> 4;      // half-wave group (0/1)
    const int ln16 = lane & 15;

    const int row0 = blockIdx.y * 128;
    const int col0 = blockIdx.x * 128;
    const int wr0  = (wave >> 1) * 32;   // 4 row-waves
    const int wc0  = (wave & 1) * 64;    // 2 col-waves

    // staging: 512 16B-chunks per tile, 2 per thread: c = t + j*256
    int r_[2], kc_[2];
    const __bf16* aP[2];
    const __bf16* bP[2];
    unsigned ldsA[2][2], ldsB[2][2];     // [buf][j]
#pragma unroll
    for (int j = 0; j < 2; ++j) {
        const int c = t + j * 256;
        r_[j]  = c >> 2;          // 0..127
        kc_[j] = (c & 3) * 8;     // 0,8,16,24
        aP[j] = A  + (size_t)(row0 + r_[j]) * K + kc_[j];
        bP[j] = Bt + (size_t)(col0 + r_[j]) * K + kc_[j];
        ldsA[0][j] = lds_off(&As[0][r_[j]][kc_[j]]);
        ldsA[1][j] = lds_off(&As[1][r_[j]][kc_[j]]);
        ldsB[0][j] = lds_off(&Bs[0][r_[j]][kc_[j]]);
        ldsB[1][j] = lds_off(&Bs[1][r_[j]][kc_[j]]);
    }

    // prologue: async-fill buffer 0
#pragma unroll
    for (int j = 0; j < 2; ++j) {
        async_b128_to_lds(ldsA[0][j], aP[j]);
        async_b128_to_lds(ldsB[0][j], bP[j]);
        aP[j] += 32; bP[j] += 32;
    }
    wait_asynccnt0();
    __syncthreads();

    v8f acc[2][4];
#pragma unroll
    for (int i = 0; i < 2; ++i)
#pragma unroll
        for (int j = 0; j < 4; ++j)
#pragma unroll
            for (int r = 0; r < 8; ++r) acc[i][j][r] = 0.0f;

    const int nk = K >> 5;
    for (int ks = 0; ks < nk; ++ks) {
        const int cur = ks & 1, nxt = cur ^ 1;
        const bool more = (ks + 1) < nk;

        // start next tile's DMA (memory -> LDS[nxt]); overlaps with WMMAs
        if (more) {
#pragma unroll
            for (int j = 0; j < 2; ++j) {
                async_b128_to_lds(ldsA[nxt][j], aP[j]);
                async_b128_to_lds(ldsB[nxt][j], bP[j]);
                aP[j] += 32; bP[j] += 32;
            }
        }

        // fragments per CDNA5 16-bit layouts
        v16bf afr[2], bfr[4];
#pragma unroll
        for (int i = 0; i < 2; ++i) {
            const int m = wr0 + i * 16 + ln16;
            const v8bf lo = *(const v8bf*)&As[cur][m][g * 8];       // K = 8g+e
            const v8bf hi = *(const v8bf*)&As[cur][m][16 + g * 8];  // K = 16+8g+(e-8)
#pragma unroll
            for (int e = 0; e < 8; ++e) { afr[i][e] = lo[e]; afr[i][8 + e] = hi[e]; }
        }
#pragma unroll
        for (int j = 0; j < 4; ++j) {
            const int n = wc0 + j * 16 + ln16;
            const v8bf lo = *(const v8bf*)&Bs[cur][n][g * 8];       // K = 8g+e
            const v8bf hi = *(const v8bf*)&Bs[cur][n][16 + g * 8];
#pragma unroll
            for (int e = 0; e < 8; ++e) { bfr[j][e] = lo[e]; bfr[j][8 + e] = hi[e]; }
        }
#pragma unroll
        for (int i = 0; i < 2; ++i)
#pragma unroll
            for (int j = 0; j < 4; ++j)
                acc[i][j] = __builtin_amdgcn_wmma_f32_16x16x32_bf16(
                    false, afr[i], false, bfr[j], (short)0, acc[i][j], false, false);

        if (more) wait_asynccnt0();   // DMA into LDS[nxt] complete
        __syncthreads();              // publish LDS[nxt] to all waves
    }

    // epilogue + store (C f32 layout: VGPR r -> row 8g+r, col lane%16)
    float*  Cf = (float*)Cv;
    __bf16* Cb = (__bf16*)Cv;
#pragma unroll
    for (int i = 0; i < 2; ++i)
#pragma unroll
        for (int j = 0; j < 4; ++j)
#pragma unroll
            for (int r = 0; r < 8; ++r) {
                const int row = row0 + wr0 + i * 16 + g * 8 + r;
                const int col = col0 + wc0 + j * 16 + ln16;
                float vv = acc[i][j][r];
                if (EPI == EPI_BIAS_ELU || EPI == EPI_BIAS_RES) vv += bias[col];
                if (EPI == EPI_BIAS_ELU) vv = (vv > 0.0f) ? vv : (__expf(vv) - 1.0f);
                if (EPI == EPI_RES || EPI == EPI_BIAS_RES)
                    vv += resid[(size_t)row * N + col];
                if (OUTB) Cb[(size_t)row * N + col] = (__bf16)vv;
                else      Cf[(size_t)row * N + col] = vv;
            }
}

// =====================================================================
// fp32 -> bf16 elementwise (x4 vectorized)
// =====================================================================
__global__ void __launch_bounds__(256)
f32_to_bf16_k(const float* __restrict__ in, __bf16* __restrict__ outp, int n)
{
    const int i = (blockIdx.x * 256 + threadIdx.x) * 4;
    if (i + 3 < n) {
        const v4f v = *(const v4f*)(in + i);
        outp[i + 0] = (__bf16)v[0];
        outp[i + 1] = (__bf16)v[1];
        outp[i + 2] = (__bf16)v[2];
        outp[i + 3] = (__bf16)v[3];
    }
}

// =====================================================================
// fp32 [K,N] -> bf16 [N,K] transpose-convert (32x32 LDS tile)
// =====================================================================
__global__ void __launch_bounds__(256)
f32_to_bf16_tr(const float* __restrict__ in, __bf16* __restrict__ outp, int K, int N)
{
    __shared__ float tile[32][33];
    const int tx = threadIdx.x & 31, ty = threadIdx.x >> 5;   // 32x8
    const int nb = blockIdx.x * 32, kb = blockIdx.y * 32;
#pragma unroll
    for (int i = ty; i < 32; i += 8)
        tile[i][tx] = in[(size_t)(kb + i) * N + nb + tx];
    __syncthreads();
#pragma unroll
    for (int i = ty; i < 32; i += 8)
        outp[(size_t)(nb + i) * K + kb + tx] = (__bf16)tile[tx][i];
}

// =====================================================================
// Performer relu random features: out[pos,m] = relu(r * q[pos,:].P[m,:]) + stab
// =====================================================================
__global__ void __launch_bounds__(256)
perf_features(const float* __restrict__ qk, const float* __restrict__ P,
              float* __restrict__ outp, int npos)
{
    __shared__ float Ps[512];
    for (int i = threadIdx.x; i < 512; i += 256) Ps[i] = P[i];
    __syncthreads();
    const int gid = blockIdx.x * 256 + threadIdx.x;
    if (gid >= npos * 8) return;
    const int pos = gid >> 3, m = gid & 7;
    const float* qr = qk + (size_t)pos * 64;
    const float* pr = Ps + m * 64;
    float acc = 0.0f;
#pragma unroll
    for (int d = 0; d < 64; ++d) acc += qr[d] * pr[d];
    acc *= 0.35355339059327373f;                 // 1/sqrt(M=8)
    outp[gid] = fmaxf(acc, 0.0f) + 0.001f;       // relu + stabilizer
}

// =====================================================================
// kv[b,h,m,d] = sum_l kp[b,l,h,m]*v[b,l,h,d]; ksum[b,h,m] = sum_l kp
// =====================================================================
__global__ void __launch_bounds__(512)
perf_kv(const float* __restrict__ kp, const float* __restrict__ v,
        float* __restrict__ kv, float* __restrict__ ksum,
        int Lt, int Ht, int chunk)
{
    const int bh = blockIdx.x, b = bh / Ht, h = bh % Ht;
    const int l0 = blockIdx.y * chunk;
    const int t = threadIdx.x, m = t >> 6, d = t & 63;
    __shared__ float kps[32][8];
    __shared__ float vs[32][64];
    float acc = 0.0f, accs = 0.0f;
    for (int lb = 0; lb < chunk; lb += 32) {
        for (int i = t; i < 32 * 8; i += 512) {
            const int il = i >> 3, im = i & 7;
            kps[il][im] = kp[(((size_t)(b * Lt + l0 + lb + il) * Ht + h) << 3) + im];
        }
        for (int i = t; i < 32 * 64; i += 512) {
            const int il = i >> 6, id = i & 63;
            vs[il][id] = v[((size_t)(b * Lt + l0 + lb + il) << 10) + (h << 6) + id];
        }
        __syncthreads();
#pragma unroll
        for (int i = 0; i < 32; ++i) {
            const float kpv = kps[i][m];
            acc += kpv * vs[i][d];
            accs += kpv;
        }
        __syncthreads();
    }
    atomicAdd(&kv[(((size_t)bh * 8 + m) << 6) + d], acc);
    if (d == 0) atomicAdd(&ksum[bh * 8 + m], accs);
}

// =====================================================================
// o[b,l,h,d] = (sum_m qp[m]*kv[m,d]) / (sum_m qp[m]*ksum[m])   (bf16 out)
// =====================================================================
__global__ void __launch_bounds__(256)
perf_out(const float* __restrict__ qp, const float* __restrict__ kv,
         const float* __restrict__ ksum, __bf16* __restrict__ o,
         int Lt, int Ht)
{
    const int gid = blockIdx.x * 256 + threadIdx.x;   // over npos*64
    const int pos = gid >> 6, d = gid & 63;
    const int h = pos % Ht, n = pos / Ht, b = n / Lt;
    const int bh = b * Ht + h;
    const float* qr  = qp + (size_t)pos * 8;
    const float* kvb = kv + ((size_t)bh * 8 << 6);
    const float* ksb = ksum + bh * 8;
    float num = 0.0f, den = 0.0f;
#pragma unroll
    for (int m = 0; m < 8; ++m) {
        const float q = qr[m];
        num += q * kvb[(m << 6) + d];
        den += q * ksb[m];
    }
    o[((size_t)n << 10) + (h << 6) + d] = (__bf16)(num / den);
}

// =====================================================================
// LayerNorm over D=1024, one block per row, optional bf16 secondary output
// =====================================================================
__global__ void __launch_bounds__(256)
layernorm_k(const float* __restrict__ x, const float* __restrict__ gamma,
            const float* __restrict__ beta, float* __restrict__ y,
            __bf16* __restrict__ yb)
{
    __shared__ float s1[256], s2[256];
    const int row = blockIdx.x;
    const float* xr = x + ((size_t)row << 10);
    float v[4], sum = 0.0f, sq = 0.0f;
#pragma unroll
    for (int j = 0; j < 4; ++j) {
        v[j] = xr[threadIdx.x + j * 256];
        sum += v[j]; sq += v[j] * v[j];
    }
    s1[threadIdx.x] = sum; s2[threadIdx.x] = sq;
    __syncthreads();
    for (int st = 128; st > 0; st >>= 1) {
        if (threadIdx.x < st) {
            s1[threadIdx.x] += s1[threadIdx.x + st];
            s2[threadIdx.x] += s2[threadIdx.x + st];
        }
        __syncthreads();
    }
    const float mean = s1[0] * (1.0f / 1024.0f);
    const float var  = s2[0] * (1.0f / 1024.0f) - mean * mean;
    const float rstd = rsqrtf(var + 1e-6f);
#pragma unroll
    for (int j = 0; j < 4; ++j) {
        const int c = threadIdx.x + j * 256;
        const float o = (v[j] - mean) * rstd * gamma[c] + beta[c];
        y[((size_t)row << 10) + c] = o;
        if (yb) yb[((size_t)row << 10) + c] = (__bf16)o;
    }
}

__global__ void zero_k(float* __restrict__ p, int n)
{
    const int i = blockIdx.x * 256 + threadIdx.x;
    if (i < n) p[i] = 0.0f;
}

// =====================================================================
// Orchestration
// =====================================================================
extern "C" void kernel_launch(void* const* d_in, const int* in_sizes, int n_in,
                              void* d_out, int out_size, void* d_ws, size_t ws_size,
                              hipStream_t stream)
{
    const float* x    = (const float*)d_in[0];
    const float* enc  = (const float*)d_in[1];
    const float* Wq1  = (const float*)d_in[2];
    const float* Wk1  = (const float*)d_in[3];
    const float* Wv1  = (const float*)d_in[4];
    const float* Wo1  = (const float*)d_in[5];
    const float* P1   = (const float*)d_in[6];
    const float* Wq2  = (const float*)d_in[7];
    const float* Wk2  = (const float*)d_in[8];
    const float* Wv2  = (const float*)d_in[9];
    const float* Wo2  = (const float*)d_in[10];
    const float* P2   = (const float*)d_in[11];
    const float* W1   = (const float*)d_in[12];
    const float* b1   = (const float*)d_in[13];
    const float* W2   = (const float*)d_in[14];
    const float* b2   = (const float*)d_in[15];
    const float* g1   = (const float*)d_in[16];
    const float* be1  = (const float*)d_in[17];
    const float* g2   = (const float*)d_in[18];
    const float* be2  = (const float*)d_in[19];
    const float* g3   = (const float*)d_in[20];
    const float* be3  = (const float*)d_in[21];
    float* out = (float*)d_out;

    const int Nrow = 4 * 4096;   // B*L
    const int D = 1024, DFF = 4096;
    const int Bc = 4, Lc = 4096, Hc = 16;

    // ---- workspace carving (all sizes multiples of 16B) ----
    char* base = (char*)d_ws;
    auto allocF = [&](size_t n) { float*  p = (float*)base;  base += n * sizeof(float);  return p; };
    auto allocB = [&](size_t n) { __bf16* p = (__bf16*)base; base += n * sizeof(__bf16); return p; };

    float* q    = allocF((size_t)Nrow * 1024);
    float* kbuf = allocF((size_t)Nrow * 1024);
    float* vbuf = allocF((size_t)Nrow * 1024);
    float* qp   = allocF((size_t)Nrow * 128);
    float* kp   = allocF((size_t)Nrow * 128);
    float* kvb  = allocF(4 * 16 * 8 * 64);        // 32768
    float* ksb  = allocF(1024);                   // 512 used
    float* out1 = allocF((size_t)Nrow * 1024);
    float* out2 = allocF((size_t)Nrow * 1024);
    float* s    = kbuf;   // k consumed before residual-sum is produced

    __bf16* xb   = allocB((size_t)Nrow * 1024);
    __bf16* encb = allocB((size_t)Nrow * 1024);
    __bf16* ob   = allocB((size_t)Nrow * 1024);   // attention output (both blocks)
    __bf16* o1b  = allocB((size_t)Nrow * 1024);   // out1 / out2 bf16 (reused)
    __bf16* ffb  = allocB((size_t)Nrow * 4096);
    __bf16* o2b  = o1b;                           // safe: o1b consumed before LN2 writes
    __bf16* wq1b = allocB((size_t)D * D);
    __bf16* wk1b = allocB((size_t)D * D);
    __bf16* wv1b = allocB((size_t)D * D);
    __bf16* wo1b = allocB((size_t)D * D);
    __bf16* wq2b = allocB((size_t)D * D);
    __bf16* wk2b = allocB((size_t)D * D);
    __bf16* wv2b = allocB((size_t)D * D);
    __bf16* wo2b = allocB((size_t)D * D);
    __bf16* w1b  = allocB((size_t)D * DFF);
    __bf16* w2b  = allocB((size_t)DFF * D);

    // ---- one-time conversions ----
    const dim3 trDD(D / 32, D / 32);
    f32_to_bf16_tr<<<trDD, 256, 0, stream>>>(Wq1, wq1b, D, D);
    f32_to_bf16_tr<<<trDD, 256, 0, stream>>>(Wk1, wk1b, D, D);
    f32_to_bf16_tr<<<trDD, 256, 0, stream>>>(Wv1, wv1b, D, D);
    f32_to_bf16_tr<<<trDD, 256, 0, stream>>>(Wo1, wo1b, D, D);
    f32_to_bf16_tr<<<trDD, 256, 0, stream>>>(Wq2, wq2b, D, D);
    f32_to_bf16_tr<<<trDD, 256, 0, stream>>>(Wk2, wk2b, D, D);
    f32_to_bf16_tr<<<trDD, 256, 0, stream>>>(Wv2, wv2b, D, D);
    f32_to_bf16_tr<<<trDD, 256, 0, stream>>>(Wo2, wo2b, D, D);
    f32_to_bf16_tr<<<dim3(DFF / 32, D / 32), 256, 0, stream>>>(W1, w1b, D, DFF);
    f32_to_bf16_tr<<<dim3(D / 32, DFF / 32), 256, 0, stream>>>(W2, w2b, DFF, D);
    const int convBlocks = (Nrow * 1024 / 4) / 256;
    f32_to_bf16_k<<<convBlocks, 256, 0, stream>>>(x, xb, Nrow * 1024);
    f32_to_bf16_k<<<convBlocks, 256, 0, stream>>>(enc, encb, Nrow * 1024);

    const dim3 gD(D / 128, Nrow / 128);       // [16384 x 1024]
    const dim3 gF(DFF / 128, Nrow / 128);     // [16384 x 4096]
    const int featBlocks = (Nrow * Hc * 8) / 256;
    const int outBlocks  = (Nrow * Hc * 64) / 256;
    const dim3 gKV(Bc * Hc, Lc / 1024);

    // -------- block 1: Performer self-attention --------
    gemm_bf16_wmma<EPI_NONE, false><<<gD, 256, 0, stream>>>(xb, wq1b, q,    nullptr, nullptr, Nrow, D, D);
    gemm_bf16_wmma<EPI_NONE, false><<<gD, 256, 0, stream>>>(xb, wk1b, kbuf, nullptr, nullptr, Nrow, D, D);
    gemm_bf16_wmma<EPI_NONE, false><<<gD, 256, 0, stream>>>(xb, wv1b, vbuf, nullptr, nullptr, Nrow, D, D);
    perf_features<<<featBlocks, 256, 0, stream>>>(q,    P1, qp, Nrow * Hc);
    perf_features<<<featBlocks, 256, 0, stream>>>(kbuf, P1, kp, Nrow * Hc);
    zero_k<<<(33792 + 255) / 256, 256, 0, stream>>>(kvb, 33792);
    perf_kv<<<gKV, 512, 0, stream>>>(kp, vbuf, kvb, ksb, Lc, Hc, 1024);
    perf_out<<<outBlocks, 256, 0, stream>>>(qp, kvb, ksb, ob, Lc, Hc);
    gemm_bf16_wmma<EPI_RES, false><<<gD, 256, 0, stream>>>(ob, wo1b, s, nullptr, x, Nrow, D, D);
    layernorm_k<<<Nrow, 256, 0, stream>>>(s, g1, be1, out1, o1b);

    // -------- block 2: Performer cross-attention (q from enc, k/v from out1) --------
    gemm_bf16_wmma<EPI_NONE, false><<<gD, 256, 0, stream>>>(encb, wq2b, q,    nullptr, nullptr, Nrow, D, D);
    gemm_bf16_wmma<EPI_NONE, false><<<gD, 256, 0, stream>>>(o1b,  wk2b, kbuf, nullptr, nullptr, Nrow, D, D);
    gemm_bf16_wmma<EPI_NONE, false><<<gD, 256, 0, stream>>>(o1b,  wv2b, vbuf, nullptr, nullptr, Nrow, D, D);
    perf_features<<<featBlocks, 256, 0, stream>>>(q,    P2, qp, Nrow * Hc);
    perf_features<<<featBlocks, 256, 0, stream>>>(kbuf, P2, kp, Nrow * Hc);
    zero_k<<<(33792 + 255) / 256, 256, 0, stream>>>(kvb, 33792);
    perf_kv<<<gKV, 512, 0, stream>>>(kp, vbuf, kvb, ksb, Lc, Hc, 1024);
    perf_out<<<outBlocks, 256, 0, stream>>>(qp, kvb, ksb, ob, Lc, Hc);
    gemm_bf16_wmma<EPI_RES, false><<<gD, 256, 0, stream>>>(ob, wo2b, s, nullptr, out1, Nrow, D, D);
    layernorm_k<<<Nrow, 256, 0, stream>>>(s, g2, be2, out2, o2b);

    // -------- block 3: FFN --------
    gemm_bf16_wmma<EPI_BIAS_ELU, true><<<gF, 256, 0, stream>>>(o2b, w1b, ffb, b1, nullptr, Nrow, DFF, D);
    gemm_bf16_wmma<EPI_BIAS_RES, false><<<gD, 256, 0, stream>>>(ffb, w2b, s, b2, out2, Nrow, D, DFF);
    layernorm_k<<<Nrow, 256, 0, stream>>>(s, g3, be3, out, (__bf16*)nullptr);
}
